// CrossAttentionOriginal_87376814670424
// MI455X (gfx1250) — compile-verified, matrix operand order checked by reading, not run on hardware
//
#include <hip/hip_runtime.h>
#include <hip/hip_bf16.h>

// ---------------- problem constants ----------------
#define BB     4
#define NN     4096
#define QD     640
#define CD     768
#define HH     8
#define DH     80
#define INNER  640      // H*DH
#define RANK   16
#define CTX    81
#define ORG    77
#define IPA    4
#define DHP    96       // DH padded to multiple of 32
#define TOKP   80       // org tokens padded to multiple of 16
#define ITOKS  16       // ipa token pad for sim tiles
#define ITOKP  32       // ipa token pad for PV contraction
#define ROWS   (BB*NN)  // 16384
#define TK     32       // GEMM K-chunk

typedef __attribute__((ext_vector_type(16))) _Float16 v16h;
typedef __attribute__((ext_vector_type(8)))  _Float16 v8h;
typedef __attribute__((ext_vector_type(4)))  _Float16 v4h;
typedef __attribute__((ext_vector_type(8)))  float    v8f;

#define WMMA_F16(A,B,C) __builtin_amdgcn_wmma_f32_16x16x32_f16(false,(A),false,(B),(short)0,(C),false,false)
#define V8F_ZERO ((v8f){0.f,0.f,0.f,0.f,0.f,0.f,0.f,0.f})

// ---------------- workspace layout (bytes, all 256B aligned) ----------------
#define OFF_WQT     ((size_t)0)                                   // 640*640 f16 (transposed)
#define OFF_WOUTT   (OFF_WQT   + (size_t)QD*INNER*2)
#define OFF_WKT     (OFF_WOUTT + (size_t)INNER*QD*2)              // 768*640 f16
#define OFF_WVT     (OFF_WKT   + (size_t)CD*INNER*2)
#define OFF_WKIT    (OFF_WVT   + (size_t)CD*INNER*2)
#define OFF_WVIT    (OFF_WKIT  + (size_t)CD*INNER*2)
#define OFF_XH      (OFF_WVIT  + (size_t)CD*INNER*2)              // 16384*640 f16
#define OFF_ATTN    (OFF_XH    + (size_t)ROWS*QD*2)               // 16384*640 f16
// ---- zeroed K/V pad region (small) ----
#define OFF_KO      (OFF_ATTN  + (size_t)ROWS*INNER*2)            // B*H*TOKP*DHP f16
#define OFF_VOT     (OFF_KO    + (size_t)BB*HH*TOKP*DHP*2)        // B*H*DH*DHP  f16
#define OFF_KI      (OFF_VOT   + (size_t)BB*HH*DH*DHP*2)          // B*H*ITOKS*DHP f16
#define OFF_VIT     (OFF_KI    + (size_t)BB*HH*ITOKS*DHP*2)       // B*H*DH*ITOKP f16
#define OFF_QP      (OFF_VIT   + (size_t)BB*HH*DH*ITOKP*2)        // 16384*(H*DHP=768) f16 (pads written by qproj)
#define OFF_END     (OFF_QP    + (size_t)ROWS*(HH*DHP)*2)
#define ZERO_BYTES  (OFF_QP - OFF_KO)                             // only K/V buffers

// ---------------- kernel: fold LoRA into weight, store transposed f16 ----------------
// outT[n*K + k] = W[k*N + n] + sum_r down[k*RANK+r]*up[r*N+n]   (scale == 1.0 exactly)
__global__ void fuseT_kernel(const float* __restrict__ W,
                             const float* __restrict__ down,
                             const float* __restrict__ up,
                             int K, int Ncol, _Float16* __restrict__ outT)
{
    int idx = blockIdx.x * blockDim.x + threadIdx.x;
    if (idx >= K * Ncol) return;
    int n = idx / K, k = idx % K;
    float v = W[(size_t)k * Ncol + n];
    if (down) {
        float s = 0.f;
        #pragma unroll
        for (int r = 0; r < RANK; ++r)
            s += down[(size_t)k * RANK + r] * up[(size_t)r * Ncol + n];
        v += s;
    }
    outT[(size_t)n * K + k] = (_Float16)v;
}

// ---------------- kernel: cast x (f32) -> f16, 4-wide ----------------
__global__ void castx_kernel(const float4* __restrict__ x, v4h* __restrict__ xh, int n4)
{
    int i = blockIdx.x * blockDim.x + threadIdx.x;
    if (i < n4) {
        float4 f = x[i];
        v4h h; h.x = (_Float16)f.x; h.y = (_Float16)f.y; h.z = (_Float16)f.z; h.w = (_Float16)f.w;
        xh[i] = h;
    }
}

// ---------------- kernel: zero padded K/V region ----------------
__global__ void zero_kernel(float4* __restrict__ p, int n4)
{
    int i = blockIdx.x * blockDim.x + threadIdx.x;
    if (i < n4) p[i] = make_float4(0.f, 0.f, 0.f, 0.f);
}

// ---------------- kernel: K/V projections (81 tokens -> tiny), 8-wide inner ----------------
__global__ void kv_kernel(const float* __restrict__ ctx,
                          const _Float16* __restrict__ WkT,  const _Float16* __restrict__ WvT,
                          const _Float16* __restrict__ WkiT, const _Float16* __restrict__ WviT,
                          _Float16* __restrict__ Ko, _Float16* __restrict__ VoT,
                          _Float16* __restrict__ Ki, _Float16* __restrict__ ViT)
{
    int idx = blockIdx.x * blockDim.x + threadIdx.x;
    if (idx >= BB * CTX * INNER) return;
    int b   = idx / (CTX * INNER);
    int r   = idx % (CTX * INNER);
    int tok = r / INNER;
    int n   = r % INNER;
    bool is_ipa = (tok >= ORG);
    const float*    cp = ctx + ((size_t)b * CTX + tok) * CD;
    const _Float16* wk = (is_ipa ? WkiT : WkT) + (size_t)n * CD;
    const _Float16* wv = (is_ipa ? WviT : WvT) + (size_t)n * CD;
    float ak = 0.f, av = 0.f;
    for (int k = 0; k < CD; k += 8) {
        float4 c0 = *(const float4*)(cp + k);
        float4 c1 = *(const float4*)(cp + k + 4);
        v8h wkv = *(const v8h*)(wk + k);
        v8h wvv = *(const v8h*)(wv + k);
        ak += c0.x*(float)wkv[0] + c0.y*(float)wkv[1] + c0.z*(float)wkv[2] + c0.w*(float)wkv[3]
            + c1.x*(float)wkv[4] + c1.y*(float)wkv[5] + c1.z*(float)wkv[6] + c1.w*(float)wkv[7];
        av += c0.x*(float)wvv[0] + c0.y*(float)wvv[1] + c0.z*(float)wvv[2] + c0.w*(float)wvv[3]
            + c1.x*(float)wvv[4] + c1.y*(float)wvv[5] + c1.z*(float)wvv[6] + c1.w*(float)wvv[7];
    }
    int h = n / DH, d = n % DH, bh = b * HH + h;
    if (!is_ipa) {
        Ko [((size_t)bh * TOKP + tok) * DHP + d  ] = (_Float16)ak;
        VoT[((size_t)bh * DH   + d  ) * DHP + tok] = (_Float16)av;
    } else {
        int it = tok - ORG;
        Ki [((size_t)bh * ITOKS + it) * DHP   + d ] = (_Float16)ak;
        ViT[((size_t)bh * DH    + d ) * ITOKP + it] = (_Float16)av;
    }
}

// ---------------- kernel: Q = x @ WqT_eff (WMMA, pipelined LDS-staged B), head-padded out ----
// grid (ROWS/128, INNER/64), block 256 (8 waves). Software pipeline: next-tile global loads
// are issued at iteration start (registers), consumed/ds_stored only AFTER the 4 WMMAs, so
// their s_wait_loadcnt sits behind a full chunk of matrix work.
__global__ void qproj_kernel(const _Float16* __restrict__ X,
                             const _Float16* __restrict__ WT,
                             _Float16* __restrict__ QP)
{
    __shared__ _Float16 Bls[2][64 * TK];   // 2 x 4KB
    int tid  = threadIdx.x;
    int lane = tid & 31, w = tid >> 5;
    int row0 = blockIdx.x * 128 + w * 16;
    int col0 = blockIdx.y * 64;
    int arow = row0 + (lane & 15);
    int koff = (lane >> 4) * 16;
    int colB = tid >> 2;                 // 0..63
    int kpB  = (tid & 3) * 8;            // 0,8,16,24
    const _Float16* wsrc = WT + (size_t)(col0 + colB) * QD + kpB;
    const _Float16* asrc = X + (size_t)arow * QD + koff;

    v8f acc[4];
    #pragma unroll
    for (int t = 0; t < 4; ++t) acc[t] = V8F_ZERO;

    // prologue: stage first B tile, load first A fragment
    *(v8h*)(Bls[0] + colB * TK + kpB) = *(const v8h*)(wsrc);
    v16h a_cur = *(const v16h*)(asrc);
    __syncthreads();

    int buf = 0;
    for (int kc = 0; kc < QD; kc += TK) {
        bool more = (kc + TK < QD);
        v8h  b_next;
        v16h a_next;
        if (more) {                       // issue next-tile loads, no wait yet
            b_next = *(const v8h*)(wsrc + kc + TK);
            a_next = *(const v16h*)(asrc + kc + TK);
            __builtin_prefetch(asrc + kc + 2 * TK, 0, 0);
        }
        #pragma unroll
        for (int t = 0; t < 4; ++t) {     // current chunk: LDS B frags + WMMA
            v16h bv = *(const v16h*)(&Bls[buf][(t * 16 + (lane & 15)) * TK + koff]);
            acc[t] = WMMA_F16(a_cur, bv, acc[t]);
        }
        if (more)                         // waits for b_next here, behind the WMMAs
            *(v8h*)(Bls[buf ^ 1] + colB * TK + kpB) = b_next;
        __syncthreads();
        buf ^= 1;
        a_cur = a_next;
    }

    int rbase = (lane >> 4) << 3;
    #pragma unroll
    for (int t = 0; t < 4; ++t) {
        int n = col0 + t * 16 + (lane & 15);
        int h = n / DH, d = n % DH;
        #pragma unroll
        for (int r = 0; r < 8; ++r) {
            int row = row0 + rbase + r;
            QP[(size_t)row * (HH * DHP) + h * DHP + d] = (_Float16)acc[t][r];
        }
    }
    // write head-pad zeros (cols 80..95 of each head) for this block's 128 rows
    if (blockIdx.y == 0) {
        v16h z = {0,0,0,0,0,0,0,0,0,0,0,0,0,0,0,0};
        #pragma unroll
        for (int s = 0; s < 4; ++s) {
            int seg = lane * 4 + s;          // 0..127 -> 16 rows x 8 heads (per wave's 16 rows)
            int r = seg >> 3, h = seg & 7;
            *(v16h*)(QP + (size_t)(row0 + r) * (HH * DHP) + h * DHP + DH) = z;
        }
    }
}

// ---------------- kernel: fused dual attention (org 77 keys + ipa 4 keys) ----------------
// grid (NN/16, HH, BB), block 32 (one wave)
__global__ void attn_kernel(const _Float16* __restrict__ QP,
                            const _Float16* __restrict__ Ko, const _Float16* __restrict__ VoT,
                            const _Float16* __restrict__ Ki, const _Float16* __restrict__ ViT,
                            _Float16* __restrict__ OUT)
{
    __shared__ float    sim [16 * TOKP];
    __shared__ float    simi[16 * ITOKS];
    __shared__ _Float16 pO  [16 * DHP];
    __shared__ _Float16 pI  [16 * ITOKP];

    const float scale = 0.1118033988749895f;  // 80^-0.5
    int lane = threadIdx.x & 31;
    int b = blockIdx.z, h = blockIdx.y;
    int row0 = blockIdx.x * 16;
    int bh = b * HH + h;
    int koff = (lane >> 4) * 16;
    int rbase = (lane >> 4) << 3;

    const _Float16* qbase = QP + (size_t)(b * NN + row0 + (lane & 15)) * (HH * DHP) + h * DHP;
    v16h aq[3];
    #pragma unroll
    for (int c = 0; c < 3; ++c) aq[c] = *(const v16h*)(qbase + c * 32 + koff);

    // sim = Q K^T * scale : 5 token tiles x 3 K-chunks
    #pragma unroll
    for (int jt = 0; jt < 5; ++jt) {
        v8f s = V8F_ZERO;
        #pragma unroll
        for (int c = 0; c < 3; ++c) {
            v16h kb = *(const v16h*)(Ko + ((size_t)bh * TOKP + jt * 16 + (lane & 15)) * DHP + c * 32 + koff);
            s = WMMA_F16(aq[c], kb, s);
        }
        #pragma unroll
        for (int r = 0; r < 8; ++r)
            sim[(rbase + r) * TOKP + jt * 16 + (lane & 15)] = s[r] * scale;
    }
    // ipa sim
    {
        v8f s = V8F_ZERO;
        #pragma unroll
        for (int c = 0; c < 3; ++c) {
            v16h kb = *(const v16h*)(Ki + ((size_t)bh * ITOKS + (lane & 15)) * DHP + c * 32 + koff);
            s = WMMA_F16(aq[c], kb, s);
        }
        #pragma unroll
        for (int r = 0; r < 8; ++r)
            simi[(rbase + r) * ITOKS + (lane & 15)] = s[r] * scale;
    }
    __syncthreads();

    // softmax: 2 lanes per row; halves combine via wave32 xor-shuffle across lane 16
    {
        int row  = lane & 15;
        int half = lane >> 4;
        int j0 = half ? 39 : 0;
        int j1 = half ? ORG : 39;
        float pm = -1e30f;
        for (int j = j0; j < j1; ++j) pm = fmaxf(pm, sim[row * TOKP + j]);
        float m = fmaxf(pm, __shfl_xor(pm, 16, 32));
        float ps = 0.f;
        for (int j = j0; j < j1; ++j) {
            float e = __expf(sim[row * TOKP + j] - m);
            sim[row * TOKP + j] = e;
            ps += e;
        }
        float sum = ps + __shfl_xor(ps, 16, 32);
        float inv = 1.f / sum;
        __syncthreads();
        int c0 = half * 48;
        for (int j = c0; j < c0 + 48; ++j)
            pO[row * DHP + j] = (j < ORG) ? (_Float16)(sim[row * TOKP + j] * inv) : (_Float16)0.f;

        // ipa softmax: 4 cols -> compute redundantly per half, write own half of pI
        float m2 = -1e30f;
        #pragma unroll
        for (int j = 0; j < IPA; ++j) m2 = fmaxf(m2, simi[row * ITOKS + j]);
        float s2 = 0.f;
        float e2[IPA];
        #pragma unroll
        for (int j = 0; j < IPA; ++j) { e2[j] = __expf(simi[row * ITOKS + j] - m2); s2 += e2[j]; }
        float inv2 = 1.f / s2;
        int ci = half * 16;
        for (int j = ci; j < ci + 16; ++j)
            pI[row * ITOKP + j] = (j < IPA) ? (_Float16)(e2[j] * inv2) : (_Float16)0.f;
    }
    __syncthreads();

    // out = P_org @ V_org + P_ipa @ V_ipa : chained WMMAs per DH tile
    #pragma unroll
    for (int nt = 0; nt < 5; ++nt) {
        v8f o = V8F_ZERO;
        #pragma unroll
        for (int c = 0; c < 3; ++c) {
            v16h pa = *(const v16h*)(pO + (lane & 15) * DHP + c * 32 + koff);
            v16h vb = *(const v16h*)(VoT + ((size_t)bh * DH + nt * 16 + (lane & 15)) * DHP + c * 32 + koff);
            o = WMMA_F16(pa, vb, o);
        }
        {
            v16h pa = *(const v16h*)(pI + (lane & 15) * ITOKP + koff);
            v16h vb = *(const v16h*)(ViT + ((size_t)bh * DH + nt * 16 + (lane & 15)) * ITOKP + koff);
            o = WMMA_F16(pa, vb, o);
        }
        #pragma unroll
        for (int r = 0; r < 8; ++r) {
            int row = row0 + rbase + r;
            int col = h * DH + nt * 16 + (lane & 15);
            OUT[(size_t)(b * NN + row) * INNER + col] = (_Float16)o[r];
        }
    }
}

// ---------------- kernel: out = attn @ WoutT_eff + bout (WMMA, pipelined, f32 out) --------
__global__ void outproj_kernel(const _Float16* __restrict__ X,
                               const _Float16* __restrict__ WT,
                               const float* __restrict__ bout,
                               float* __restrict__ OUT)
{
    __shared__ _Float16 Bls[2][64 * TK];
    int tid  = threadIdx.x;
    int lane = tid & 31, w = tid >> 5;
    int row0 = blockIdx.x * 128 + w * 16;
    int col0 = blockIdx.y * 64;
    int arow = row0 + (lane & 15);
    int koff = (lane >> 4) * 16;
    int colB = tid >> 2;
    int kpB  = (tid & 3) * 8;
    const _Float16* wsrc = WT + (size_t)(col0 + colB) * INNER + kpB;
    const _Float16* asrc = X + (size_t)arow * INNER + koff;

    v8f acc[4];
    #pragma unroll
    for (int t = 0; t < 4; ++t) acc[t] = V8F_ZERO;

    *(v8h*)(Bls[0] + colB * TK + kpB) = *(const v8h*)(wsrc);
    v16h a_cur = *(const v16h*)(asrc);
    __syncthreads();

    int buf = 0;
    for (int kc = 0; kc < INNER; kc += TK) {
        bool more = (kc + TK < INNER);
        v8h  b_next;
        v16h a_next;
        if (more) {
            b_next = *(const v8h*)(wsrc + kc + TK);
            a_next = *(const v16h*)(asrc + kc + TK);
            __builtin_prefetch(asrc + kc + 2 * TK, 0, 0);
        }
        #pragma unroll
        for (int t = 0; t < 4; ++t) {
            v16h bv = *(const v16h*)(&Bls[buf][(t * 16 + (lane & 15)) * TK + koff]);
            acc[t] = WMMA_F16(a_cur, bv, acc[t]);
        }
        if (more)
            *(v8h*)(Bls[buf ^ 1] + colB * TK + kpB) = b_next;
        __syncthreads();
        buf ^= 1;
        a_cur = a_next;
    }

    int rbase = (lane >> 4) << 3;
    #pragma unroll
    for (int t = 0; t < 4; ++t) {
        int n = col0 + t * 16 + (lane & 15);
        float bb = bout[n];
        #pragma unroll
        for (int r = 0; r < 8; ++r) {
            int row = row0 + rbase + r;
            OUT[(size_t)row * QD + n] = acc[t][r] + bb;
        }
    }
}

// ---------------- launcher ----------------
extern "C" void kernel_launch(void* const* d_in, const int* in_sizes, int n_in,
                              void* d_out, int out_size, void* d_ws, size_t ws_size,
                              hipStream_t stream)
{
    (void)in_sizes; (void)n_in; (void)out_size; (void)ws_size;
    const float* x       = (const float*)d_in[0];
    const float* context = (const float*)d_in[1];
    const float* Wq      = (const float*)d_in[2];
    const float* Wk      = (const float*)d_in[3];
    const float* Wv      = (const float*)d_in[4];
    const float* Wout    = (const float*)d_in[5];
    const float* bout    = (const float*)d_in[6];
    const float* q_down  = (const float*)d_in[7];
    const float* q_up    = (const float*)d_in[8];
    const float* k_down  = (const float*)d_in[10];
    const float* k_up    = (const float*)d_in[11];
    const float* v_down  = (const float*)d_in[13];
    const float* v_up    = (const float*)d_in[14];
    const float* o_down  = (const float*)d_in[16];
    const float* o_up    = (const float*)d_in[17];
    const float* Wk_ipa  = (const float*)d_in[19];
    const float* Wv_ipa  = (const float*)d_in[20];
    float* out = (float*)d_out;

    char* ws = (char*)d_ws;
    _Float16* WqT   = (_Float16*)(ws + OFF_WQT);
    _Float16* WoutT = (_Float16*)(ws + OFF_WOUTT);
    _Float16* WkT   = (_Float16*)(ws + OFF_WKT);
    _Float16* WvT   = (_Float16*)(ws + OFF_WVT);
    _Float16* WkiT  = (_Float16*)(ws + OFF_WKIT);
    _Float16* WviT  = (_Float16*)(ws + OFF_WVIT);
    _Float16* XH    = (_Float16*)(ws + OFF_XH);
    _Float16* ATT   = (_Float16*)(ws + OFF_ATTN);
    _Float16* Ko    = (_Float16*)(ws + OFF_KO);
    _Float16* VoT   = (_Float16*)(ws + OFF_VOT);
    _Float16* Ki    = (_Float16*)(ws + OFF_KI);
    _Float16* ViT   = (_Float16*)(ws + OFF_VIT);
    _Float16* QP    = (_Float16*)(ws + OFF_QP);

    // 1) fold LoRA into weights, store transposed f16
    {
        int n1 = QD * INNER, n2 = CD * INNER;
        fuseT_kernel<<<(n1 + 255) / 256, 256, 0, stream>>>(Wq,   q_down, q_up, QD,    INNER, WqT);
        fuseT_kernel<<<(n2 + 255) / 256, 256, 0, stream>>>(Wk,   k_down, k_up, CD,    INNER, WkT);
        fuseT_kernel<<<(n2 + 255) / 256, 256, 0, stream>>>(Wv,   v_down, v_up, CD,    INNER, WvT);
        fuseT_kernel<<<(n1 + 255) / 256, 256, 0, stream>>>(Wout, o_down, o_up, INNER, QD,    WoutT);
        fuseT_kernel<<<(n2 + 255) / 256, 256, 0, stream>>>(Wk_ipa, nullptr, nullptr, CD, INNER, WkiT);
        fuseT_kernel<<<(n2 + 255) / 256, 256, 0, stream>>>(Wv_ipa, nullptr, nullptr, CD, INNER, WviT);
    }
    // 2) cast x to f16 (vectorized); zero small K/V pad region
    {
        int n4 = ROWS * QD / 4;
        castx_kernel<<<(n4 + 255) / 256, 256, 0, stream>>>((const float4*)x, (v4h*)XH, n4);
        int z4 = (int)(ZERO_BYTES / 16);
        zero_kernel<<<(z4 + 255) / 256, 256, 0, stream>>>((float4*)(ws + OFF_KO), z4);
    }
    // 3) K/V projections (tiny)
    {
        int nkv = BB * CTX * INNER;
        kv_kernel<<<(nkv + 255) / 256, 256, 0, stream>>>(context, WkT, WvT, WkiT, WviT,
                                                         Ko, VoT, Ki, ViT);
    }
    // 4) Q projection (WMMA, pipelined LDS weights; also writes head-pad zeros)
    qproj_kernel<<<dim3(ROWS / 128, INNER / 64), 256, 0, stream>>>(XH, WqT, QP);
    // 5) fused dual attention (WMMA)
    attn_kernel<<<dim3(NN / 16, HH, BB), 32, 0, stream>>>(QP, Ko, VoT, Ki, ViT, ATT);
    // 6) output projection + bias (WMMA)
    outproj_kernel<<<dim3(ROWS / 128, QD / 64), 256, 0, stream>>>(ATT, WoutT, bout, out);
}
